// AttentionalSpikingSSMLayer_66529043415049
// MI455X (gfx1250) — compile-verified, matrix operand order checked by reading, not run on hardware
//
#include <hip/hip_runtime.h>
#include <math.h>

// ---------------------------------------------------------------------------
// AttentionalSpikingSSMLayer for MI455X (gfx1250, wave32, WMMA).
// B=4, T=96, S=128, dm=512, ds=256, H=4, dh=64, BS=512.
// Per-timestep kernel DAG (~12 launches), 96 steps enqueued on `stream`.
// All GEMMs use v_wmma_f32_16x16x32_bf16 (bf16 in, f32 accumulate).
// Each wave computes a 16x64 output strip: one A fragment feeds 4 WMMAs.
// ---------------------------------------------------------------------------

typedef __bf16 bf16;
typedef __attribute__((ext_vector_type(16))) __bf16 v16bf;
typedef __attribute__((ext_vector_type(8)))  float  v8f;

#define BS_   512
#define DS_   256
#define DM_   512
#define DH_   64
#define T_    96
#define S_    128
#define DECAYF 0.60653065971263342f   /* exp(-1/2) */
#define ADAPTF 0.1f
#define TGT_RATE 0.1f

#define WMMA_BF16(a, b, c) \
  __builtin_amdgcn_wmma_f32_16x16x32_bf16(false, (a), false, (b), (short)0, (c), false, false)

// k-offset inside a 16-bit WMMA fragment (ISA 7.12.2, 16-bit A 16x32 layout):
// lanes 0-15 (half=0): V0..3 -> k=0..7, V4..7 -> k=16..23
// lanes16-31 (half=1): V0..3 -> k=8..15, V4..7 -> k=24..31
__device__ __forceinline__ int frag_k(int j, int half) {
  return (j < 4 ? 2 * j : 16 + 2 * (j - 4)) + 8 * half;
}

// Fragment whose k-dimension is contiguous in memory.
// For A (row-major [M,K]): p = &A[m0 + (lane&15)][k0].
// For B = W^T with W row-major [N,K]: p = &W[n0 + (lane&15)][k0].
__device__ __forceinline__ v16bf load_frag_contig(const bf16* p) {
  v16bf f;
  const int half = (threadIdx.x & 31) >> 4;
#pragma unroll
  for (int j = 0; j < 8; ++j) {
    const int kk = frag_k(j, half);
    f[2 * j]     = p[kk];
    f[2 * j + 1] = p[kk + 1];
  }
  return f;
}

// Fragment whose k-dimension is strided (B from row-major [K,N] matrix):
// p = &Mkn[k0][n0 + (lane&15)], ld = row stride in elements.
__device__ __forceinline__ v16bf load_frag_strided(const bf16* p, int ld) {
  v16bf f;
  const int half = (threadIdx.x & 31) >> 4;
#pragma unroll
  for (int j = 0; j < 8; ++j) {
    const int kk = frag_k(j, half);
    f[2 * j]     = p[(size_t)kk * ld];
    f[2 * j + 1] = p[(size_t)(kk + 1) * ld];
  }
  return f;
}

// Write one 16x16 accumulator tile (f32 and/or bf16, optional bias/addsrc).
__device__ __forceinline__ void store_tile(v8f acc, int m0, int col, int N,
                                           const float* __restrict__ bias,
                                           const float* __restrict__ addsrc,
                                           float* __restrict__ outf,
                                           bf16*  __restrict__ outb) {
  const int half = (threadIdx.x & 31) >> 4;
  const float bb = bias ? bias[col] : 0.0f;
#pragma unroll
  for (int r = 0; r < 8; ++r) {
    const int row = m0 + r + 8 * half;
    float v = acc[r] + bb;
    if (addsrc) v += addsrc[(size_t)row * N + col];
    if (outf) outf[(size_t)row * N + col] = v;
    if (outb) outb[(size_t)row * N + col] = (bf16)v;
  }
}

// ---------------------------------------------------------------------------
// Generic Y = A @ W^T (+bias) (+addsrc) GEMM. One wave per 16x64 output
// strip (4 accumulators share one A fragment per k-step -> 4 back-to-back
// WMMAs). 4 waves / block. M mult of 16, N mult of 64, K mult of 32.
// Optional device-flag guard (wave-uniform exit: EXEC==all-ones at WMMA).
// ---------------------------------------------------------------------------
__global__ void gemm_bf16_kernel(const bf16* __restrict__ A, int lda,
                                 const bf16* __restrict__ W, int ldw,
                                 const float* __restrict__ bias,
                                 const float* __restrict__ addsrc,
                                 float* __restrict__ outf,
                                 bf16*  __restrict__ outb,
                                 int M, int N, int K,
                                 const int* __restrict__ flag) {
  if (flag && *flag == 0) return;
  const int lane = threadIdx.x & 31;
  const int wave = threadIdx.x >> 5;
  const int stripsN = N >> 6;                // 64-wide strips
  const int strip = blockIdx.x * 4 + wave;
  if (strip >= (M >> 4) * stripsN) return;
  const int m0 = (strip / stripsN) << 4;
  const int n0 = (strip % stripsN) << 6;

  v8f acc0 = {}, acc1 = {}, acc2 = {}, acc3 = {};
  const bf16* ap  = A + (size_t)(m0 + (lane & 15)) * lda;
  const bf16* wp0 = W + (size_t)(n0 +      (lane & 15)) * ldw;
  const bf16* wp1 = W + (size_t)(n0 + 16 + (lane & 15)) * ldw;
  const bf16* wp2 = W + (size_t)(n0 + 32 + (lane & 15)) * ldw;
  const bf16* wp3 = W + (size_t)(n0 + 48 + (lane & 15)) * ldw;

  for (int k0 = 0; k0 < K; k0 += 32) {
    // Prefetch next k-chunk (lowers to global_prefetch_b8; L2 stays warm).
    if (k0 + 32 < K) {
      __builtin_prefetch(ap + k0 + 32, 0, 0);
      __builtin_prefetch(wp0 + k0 + 32, 0, 0);
    }
    v16bf a  = load_frag_contig(ap + k0);
    v16bf b0 = load_frag_contig(wp0 + k0);
    v16bf b1 = load_frag_contig(wp1 + k0);
    v16bf b2 = load_frag_contig(wp2 + k0);
    v16bf b3 = load_frag_contig(wp3 + k0);
    acc0 = WMMA_BF16(a, b0, acc0);
    acc1 = WMMA_BF16(a, b1, acc1);
    acc2 = WMMA_BF16(a, b2, acc2);
    acc3 = WMMA_BF16(a, b3, acc3);
  }
  store_tile(acc0, m0, n0 +      (lane & 15), N, bias, addsrc, outf, outb);
  store_tile(acc1, m0, n0 + 16 + (lane & 15), N, bias, addsrc, outf, outb);
  store_tile(acc2, m0, n0 + 32 + (lane & 15), N, bias, addsrc, outf, outb);
  store_tile(acc3, m0, n0 + 48 + (lane & 15), N, bias, addsrc, outf, outb);
}

// ---------------------------------------------------------------------------
// Fused attention: scores = softmax((Q Kᵀ)/8) per head, av = P V.
// grid = (32 query-tiles, 4 heads), block = 128 (4 waves).
// Scores staged in LDS (16x512 f32 + 16x512 bf16 = 48 KB).
// Q fragments (dh=64 -> two v16bf) are hoisted and live across phase 1.
// ---------------------------------------------------------------------------
__global__ void attn_kernel(const bf16* __restrict__ qb,
                            const bf16* __restrict__ kb,
                            const bf16* __restrict__ vb,
                            bf16* __restrict__ avb,
                            const int* __restrict__ flag) {
  if (*flag == 0) return;
  __shared__ float sS[16 * BS_];
  __shared__ bf16  sP[16 * BS_];
  const int head = blockIdx.y;
  const int m0   = blockIdx.x << 4;
  const int lane = threadIdx.x & 31;
  const int wave = threadIdx.x >> 5;
  const int half = lane >> 4;
  const float scale = 0.125f;  // 1/sqrt(64)

  const bf16* qp = qb + (size_t)(m0 + (lane & 15)) * DS_ + head * DH_;
  const v16bf qf0 = load_frag_contig(qp);
  const v16bf qf1 = load_frag_contig(qp + 32);

  // Phase 1: scores for this 16-query strip against all 512 keys.
  for (int nt = wave; nt < BS_ / 16; nt += 4) {
    const bf16* kp = kb + (size_t)(nt * 16 + (lane & 15)) * DS_ + head * DH_;
    v8f acc = {};
    acc = WMMA_BF16(qf0, load_frag_contig(kp),      acc);
    acc = WMMA_BF16(qf1, load_frag_contig(kp + 32), acc);
    const int n = nt * 16 + (lane & 15);
#pragma unroll
    for (int r = 0; r < 8; ++r) sS[(r + 8 * half) * BS_ + n] = acc[r] * scale;
  }
  __syncthreads();

  // Phase 2: row-wise stable softmax; result as bf16 probabilities in LDS.
  if (threadIdx.x < 16) {
    const int m = threadIdx.x;
    float mx = -3.4e38f;
    for (int n = 0; n < BS_; ++n) mx = fmaxf(mx, sS[m * BS_ + n]);
    float sum = 0.f;
    for (int n = 0; n < BS_; ++n) {
      const float e = __expf(sS[m * BS_ + n] - mx);
      sS[m * BS_ + n] = e;
      sum += e;
    }
    const float inv = 1.0f / sum;
    for (int n = 0; n < BS_; ++n) sP[m * BS_ + n] = (bf16)(sS[m * BS_ + n] * inv);
  }
  __syncthreads();

  // Phase 3: av[16, 64] = P[16,512] @ V[512,64]; one wave per 16-wide d-tile.
  const int nt = wave;  // 0..3
  v8f acc = {};
  const bf16* pp = sP + (size_t)(lane & 15) * BS_;
  for (int k0 = 0; k0 < BS_; k0 += 32) {
    v16bf a = load_frag_contig(pp + k0);
    v16bf b = load_frag_strided(
        vb + (size_t)k0 * DS_ + head * DH_ + nt * 16 + (lane & 15), DS_);
    acc = WMMA_BF16(a, b, acc);
  }
#pragma unroll
  for (int r = 0; r < 8; ++r)
    avb[(size_t)(m0 + r + 8 * half) * DS_ + head * DH_ + nt * 16 + (lane & 15)] =
        (bf16)acc[r];
}

// ---------------------------------------------------------------------------
// Adaptive LIF. One thread per feature column -> threshold update is a local
// reduction over the 512 rows (no atomics). Spikes are exactly {0,1}: stored
// directly as bf16 for the next-step WMMA operands.
// ---------------------------------------------------------------------------
__global__ void lif1_kernel(const float* __restrict__ u, float* __restrict__ sv,
                            float* __restrict__ ts, bf16* __restrict__ hbf) {
  const int col = threadIdx.x;  // 256 threads
  const float thr = ts[col];
  float cnt = 0.f;
  for (int row = 0; row < BS_; ++row) {
    const size_t i = (size_t)row * DS_ + col;
    const float vp = sv[i] * DECAYF + u[i];
    const float sp = (vp - thr >= 0.f) ? 1.f : 0.f;
    hbf[i] = (bf16)sp;
    sv[i]  = vp * (1.f - sp);
    cnt += sp;
  }
  ts[col] = thr + ADAPTF * (cnt * (1.f / BS_) - TGT_RATE);
}

__global__ void lif2_kernel(const float* __restrict__ u2, float* __restrict__ ov,
                            float* __restrict__ to, float* __restrict__ out,
                            int t, const int* __restrict__ flag) {
  const int col = threadIdx.x;  // 512 threads
  if (*flag) {
    const float thr = to[col];
    float cnt = 0.f;
    for (int row = 0; row < BS_; ++row) {
      const int b = row >> 7, s = row & 127;
      const size_t oidx = ((size_t)(b * T_ + t) * S_ + s) * DM_ + col;
      const size_t i = (size_t)row * DM_ + col;
      const float vp = ov[i] * DECAYF + u2[i];
      const float sp = (vp - thr >= 0.f) ? 1.f : 0.f;
      out[oidx] = sp;
      ov[i] = vp * (1.f - sp);
      cnt += sp;
    }
    to[col] = thr + ADAPTF * (cnt * (1.f / BS_) - TGT_RATE);
  } else {
    for (int row = 0; row < BS_; ++row) {
      const int b = row >> 7, s = row & 127;
      out[((size_t)(b * T_ + t) * S_ + s) * DM_ + col] = 0.f;
    }
  }
}

// ---------------------------------------------------------------------------
// Helpers: flag reset, x_t gather+bf16-convert+any(x>0), f32->bf16, state init
// ---------------------------------------------------------------------------
__global__ void reset_flag_kernel(int* flag) {
  if (threadIdx.x == 0 && blockIdx.x == 0) *flag = 0;
}

__global__ void xt_convert_kernel(const float* __restrict__ x,
                                  bf16* __restrict__ xbf, int t,
                                  int* __restrict__ flag) {
  __shared__ int any;
  if (threadIdx.x == 0) any = 0;
  __syncthreads();
  int myany = 0;
  for (int i = blockIdx.x * blockDim.x + threadIdx.x; i < BS_ * DM_;
       i += gridDim.x * blockDim.x) {
    const int row = i >> 9, d = i & 511;
    const int b = row >> 7, s = row & 127;
    const float v = x[((size_t)(b * T_ + t) * S_ + s) * DM_ + d];
    xbf[i] = (bf16)v;
    if (v > 0.f) myany = 1;
  }
  if (myany) any = 1;  // benign race: all writers store 1
  __syncthreads();
  if (threadIdx.x == 0 && any) atomicOr(flag, 1);
}

__global__ void f2bf_kernel(const float* __restrict__ src,
                            bf16* __restrict__ dst, int n) {
  for (int i = blockIdx.x * blockDim.x + threadIdx.x; i < n;
       i += gridDim.x * blockDim.x)
    dst[i] = (bf16)src[i];
}

__global__ void init_state_kernel(bf16* hbf, float* sv, float* ov, float* ts,
                                  float* to) {
  const int i0 = blockIdx.x * blockDim.x + threadIdx.x;
  const int stride = gridDim.x * blockDim.x;
  for (int i = i0; i < BS_ * DS_; i += stride) { hbf[i] = (bf16)0.f; sv[i] = 0.f; }
  for (int i = i0; i < BS_ * DM_; i += stride) ov[i] = 0.f;
  for (int i = i0; i < DS_; i += stride) ts[i] = 1.0f;
  for (int i = i0; i < DM_; i += stride) to[i] = 1.0f;
}

// ---------------------------------------------------------------------------
// Host side
// ---------------------------------------------------------------------------
extern "C" void kernel_launch(void* const* d_in, const int* in_sizes, int n_in,
                              void* d_out, int out_size, void* d_ws,
                              size_t ws_size, hipStream_t stream) {
  (void)in_sizes; (void)n_in; (void)out_size; (void)ws_size;
  const float* x  = (const float*)d_in[0];
  const float* A  = (const float*)d_in[1];
  const float* C  = (const float*)d_in[2];
  const float* D  = (const float*)d_in[3];
  const float* Wq = (const float*)d_in[4];
  const float* bq = (const float*)d_in[5];
  const float* Wk = (const float*)d_in[6];
  const float* bk = (const float*)d_in[7];
  const float* Wv = (const float*)d_in[8];
  const float* bv = (const float*)d_in[9];
  const float* Wo = (const float*)d_in[10];
  const float* bo = (const float*)d_in[11];
  float* out = (float*)d_out;

  // Workspace bump allocator (deterministic layout, ~6.8 MB total).
  char* wsb = (char*)d_ws;
  size_t o = 0;
  auto alloc = [&](size_t bytes) -> char* {
    char* p = wsb + o;
    o = (o + bytes + 255) & ~(size_t)255;
    return p;
  };
  int*  flag = (int*)alloc(sizeof(int));
  bf16* Abf  = (bf16*)alloc((size_t)DS_ * DS_ * 2);
  bf16* Cbf  = (bf16*)alloc((size_t)DM_ * DS_ * 2);
  bf16* Dbf  = (bf16*)alloc((size_t)DM_ * DM_ * 2);
  bf16* Wqbf = (bf16*)alloc((size_t)DS_ * DS_ * 2);
  bf16* Wkbf = (bf16*)alloc((size_t)DS_ * DM_ * 2);
  bf16* Wvbf = (bf16*)alloc((size_t)DS_ * DM_ * 2);
  bf16* Wobf = (bf16*)alloc((size_t)DS_ * DS_ * 2);
  bf16* xbf  = (bf16*)alloc((size_t)BS_ * DM_ * 2);
  bf16* hbf  = (bf16*)alloc((size_t)BS_ * DS_ * 2);
  bf16* qbf  = (bf16*)alloc((size_t)BS_ * DS_ * 2);
  bf16* kbf  = (bf16*)alloc((size_t)BS_ * DS_ * 2);
  bf16* vbf  = (bf16*)alloc((size_t)BS_ * DS_ * 2);
  bf16* avbf = (bf16*)alloc((size_t)BS_ * DS_ * 2);
  float* sv  = (float*)alloc((size_t)BS_ * DS_ * 4);
  float* ov  = (float*)alloc((size_t)BS_ * DM_ * 4);
  float* ts  = (float*)alloc((size_t)DS_ * 4);
  float* to  = (float*)alloc((size_t)DM_ * 4);
  float* u   = (float*)alloc((size_t)BS_ * DS_ * 4);   // st, then st+gi
  float* u2  = (float*)alloc((size_t)BS_ * DM_ * 4);   // out_up

  // State init (runs every call: kernel_launch must be deterministic).
  hipLaunchKernelGGL(init_state_kernel, dim3(512), dim3(256), 0, stream,
                     hbf, sv, ov, ts, to);
  // Weights -> bf16.
  hipLaunchKernelGGL(f2bf_kernel, dim3(128), dim3(256), 0, stream, A,  Abf,  DS_ * DS_);
  hipLaunchKernelGGL(f2bf_kernel, dim3(256), dim3(256), 0, stream, C,  Cbf,  DM_ * DS_);
  hipLaunchKernelGGL(f2bf_kernel, dim3(512), dim3(256), 0, stream, D,  Dbf,  DM_ * DM_);
  hipLaunchKernelGGL(f2bf_kernel, dim3(128), dim3(256), 0, stream, Wq, Wqbf, DS_ * DS_);
  hipLaunchKernelGGL(f2bf_kernel, dim3(256), dim3(256), 0, stream, Wk, Wkbf, DS_ * DM_);
  hipLaunchKernelGGL(f2bf_kernel, dim3(256), dim3(256), 0, stream, Wv, Wvbf, DS_ * DM_);
  hipLaunchKernelGGL(f2bf_kernel, dim3(128), dim3(256), 0, stream, Wo, Wobf, DS_ * DS_);

  // Blocks = (M/16)*(N/64)/4 (4 waves per block, one 16x64 strip per wave).
  const int gemm256 = (BS_ / 16) * (DS_ / 64) / 4;  // 32 blocks
  const int gemm512 = (BS_ / 16) * (DM_ / 64) / 4;  // 64 blocks

  for (int t = 0; t < T_; ++t) {
    hipLaunchKernelGGL(reset_flag_kernel, dim3(1), dim3(32), 0, stream, flag);
    hipLaunchKernelGGL(xt_convert_kernel, dim3(256), dim3(256), 0, stream,
                       x, xbf, t, flag);
    // st = h @ Aᵀ  (always)
    hipLaunchKernelGGL(gemm_bf16_kernel, dim3(gemm256), dim3(128), 0, stream,
                       hbf, DS_, Abf, DS_, (const float*)nullptr,
                       (const float*)nullptr, u, (bf16*)nullptr,
                       BS_, DS_, DS_, (const int*)nullptr);
    // q/k/v projections (active only)
    hipLaunchKernelGGL(gemm_bf16_kernel, dim3(gemm256), dim3(128), 0, stream,
                       hbf, DS_, Wqbf, DS_, bq, (const float*)nullptr,
                       (float*)nullptr, qbf, BS_, DS_, DS_, flag);
    hipLaunchKernelGGL(gemm_bf16_kernel, dim3(gemm256), dim3(128), 0, stream,
                       xbf, DM_, Wkbf, DM_, bk, (const float*)nullptr,
                       (float*)nullptr, kbf, BS_, DS_, DM_, flag);
    hipLaunchKernelGGL(gemm_bf16_kernel, dim3(gemm256), dim3(128), 0, stream,
                       xbf, DM_, Wvbf, DM_, bv, (const float*)nullptr,
                       (float*)nullptr, vbf, BS_, DS_, DM_, flag);
    // fused attention (active only)
    hipLaunchKernelGGL(attn_kernel, dim3(BS_ / 16, 4), dim3(128), 0, stream,
                       qbf, kbf, vbf, avbf, flag);
    // u = st + av @ Woᵀ + bo (active only; u stays st when inactive)
    hipLaunchKernelGGL(gemm_bf16_kernel, dim3(gemm256), dim3(128), 0, stream,
                       avbf, DS_, Wobf, DS_, bo, u, u, (bf16*)nullptr,
                       BS_, DS_, DS_, flag);
    // LIF 1 (both paths)
    hipLaunchKernelGGL(lif1_kernel, dim3(1), dim3(DS_), 0, stream,
                       u, sv, ts, hbf);
    // u2 = h2 @ Cᵀ + x_t @ Dᵀ (active only)
    hipLaunchKernelGGL(gemm_bf16_kernel, dim3(gemm512), dim3(128), 0, stream,
                       hbf, DS_, Cbf, DS_, (const float*)nullptr,
                       (const float*)nullptr, u2, (bf16*)nullptr,
                       BS_, DM_, DS_, flag);
    hipLaunchKernelGGL(gemm_bf16_kernel, dim3(gemm512), dim3(128), 0, stream,
                       xbf, DM_, Dbf, DM_, (const float*)nullptr, u2, u2,
                       (bf16*)nullptr, BS_, DM_, DM_, flag);
    // LIF 2 + output write (zeros when inactive)
    hipLaunchKernelGGL(lif2_kernel, dim3(1), dim3(DM_), 0, stream,
                       u2, ov, to, out, t, flag);
  }
}
